// einnet_65326452572840
// MI455X (gfx1250) — compile-verified, matrix operand order checked by reading, not run on hardware
//
#include <hip/hip_runtime.h>
#include <hip/hip_bf16.h>

// ---------------------------------------------------------------------------
// CDNA5 (gfx1250): bf16 WMMA GEMMs, fragment-major packed weights, async-LDS
// double-buffered pipeline, padded-to-256 channels + equivariant pooling.
// ---------------------------------------------------------------------------

typedef __attribute__((ext_vector_type(16))) __bf16 bf16x16;
typedef __attribute__((ext_vector_type(8)))  float  f32x8;

// Pointee type for the async-LDS builtin (matches hipcc's signature:
// int __attribute__((vector_size(16))) in AS1 / AS3).
typedef int v4i __attribute__((vector_size(16)));
typedef __attribute__((address_space(1))) v4i v4i_g;
typedef __attribute__((address_space(3))) v4i v4i_l;

#if __has_builtin(__builtin_amdgcn_global_load_async_to_lds_b128)
#define ASYNC_COPY 1
#else
#define ASYNC_COPY 0
#endif

union frag_u {
    uint4   u4[2];
    bf16x16 v;
};

__device__ __forceinline__ __bf16 f2bf(float f) { return (__bf16)f; }

__device__ __forceinline__ void copy16_g2l(const __bf16* src, __bf16* dst)
{
#if ASYNC_COPY
    __builtin_amdgcn_global_load_async_to_lds_b128(
        (v4i_g*)(uintptr_t)src,
        (v4i_l*)(unsigned int)(uintptr_t)dst, 0, 0);
#else
    *(uint4*)dst = *(const uint4*)src;
#endif
}

__device__ __forceinline__ void lds_stage_fence()
{
#if ASYNC_COPY && __has_builtin(__builtin_amdgcn_s_wait_asynccnt)
    __builtin_amdgcn_s_wait_asynccnt(0);
#endif
    __syncthreads();
}

#define TILE_M 256
#define TILE_N 64
#define TILE_K 32
#define CHP    256   // padded channel stride for 240-channel tensors

// ---------------------------------------------------------------------------
// fp32 -> bf16 conversion (input activations)
// ---------------------------------------------------------------------------
__global__ __launch_bounds__(256) void f32_to_bf16_kernel(
    const float* __restrict__ src, __bf16* __restrict__ dst, size_t n)
{
    size_t i = (size_t)blockIdx.x * blockDim.x + threadIdx.x;
    if (i < n) dst[i] = f2bf(src[i]);
}

// ---------------------------------------------------------------------------
// Bias pad: out[0..np) = (i < n) ? bias[i] : 0
// ---------------------------------------------------------------------------
__global__ __launch_bounds__(256) void pad_bias_kernel(
    const float* __restrict__ b, float* __restrict__ out, int n, int np)
{
    const int i = blockIdx.x * 256 + threadIdx.x;
    if (i < np) out[i] = (i < n) ? b[i] : 0.0f;
}

// ---------------------------------------------------------------------------
// Weight repack: W (Ktrue x Ntrue, f32 row-major) -> fragment-major bf16,
// zero-padded to (32*kTiles) x Npad.
// Wp[((kt*NT + nt)*32 + lane)*16 + e] is lane `lane`'s B-fragment element
// `e` for WMMA tile (kt, nt):
//   e in 0..7  -> K = kt*32 + kSel + e        (kSel = (lane>>4)*8)
//   e in 8..15 -> K = kt*32 + kSel + 16 + e-8
//   N = nt*16 + (lane & 15)
// ---------------------------------------------------------------------------
__global__ __launch_bounds__(256) void repack_w_kernel(
    const float* __restrict__ W, __bf16* __restrict__ Wp,
    int Ktrue, int Ntrue, int Npad, int kTiles)
{
    const int NT = Npad / 16;
    const size_t total = (size_t)kTiles * NT * 32 * 16;
    const size_t i = (size_t)blockIdx.x * blockDim.x + threadIdx.x;
    if (i >= total) return;
    const int e    = (int)(i & 15);
    const int lane = (int)((i >> 4) & 31);
    const size_t tile = i >> 9;            // kt*NT + nt
    const int nt = (int)(tile % NT);
    const int kt = (int)(tile / NT);
    const int kSel = (lane >> 4) * 8;
    const int k = kt * 32 + kSel + (e >> 3) * 16 + (e & 7);
    const int n = nt * 16 + (lane & 15);
    Wp[i] = (k < Ktrue && n < Ntrue) ? f2bf(W[(size_t)k * Ntrue + n])
                                     : f2bf(0.0f);
}

// ---------------------------------------------------------------------------
// GEMM: C[M x N] = epilogue(A[M x K] * W[K x N] + bias)
// K, N compile-time, both fully tiled (K % 32 == 0, N % 64 == 0).
// A bf16 row-major (stride K); Wp fragment-major packed bf16.
// 256 threads = 8 waves; block tile 256x64; each wave owns 32 rows
// (2 A-fragments) x 4 WMMA col tiles -> 8 v_wmma per K-step.
// Double-buffered async-LDS staging: next A tile DMA overlaps WMMA.
// EPI: 0 = none, 1 = relu, 2 = sin.  OUTBF: store bf16 instead of f32.
// ---------------------------------------------------------------------------
template <int K, int N, int EPI, bool OUTBF>
__global__ __launch_bounds__(256) void gemm_wmma_kernel(
    const __bf16* __restrict__ A, const __bf16* __restrict__ Wp,
    const float* __restrict__ bias, void* __restrict__ Cout)
{
    constexpr int KT = K / TILE_K;
    constexpr int NT = N / 16;

    __shared__ __bf16 As[2][TILE_M][TILE_K + 8];   // 2 x 256 x 40 (40 KB)

    const int tid     = threadIdx.x;
    const int wave    = tid >> 5;
    const int lane    = tid & 31;
    const int l15     = lane & 15;
    const int kSel    = (lane >> 4) * 8;    // lanes 0-15: K{0..7,16..23}
                                            // lanes16-31: K{8..15,24..31}
    const int rowBase = blockIdx.x * TILE_M;
    const int ntBase  = blockIdx.y * (TILE_N / 16);

    const int rStage  = tid >> 2;           // 0..63
    const int cStage  = (tid & 3) * 8;      // 0,8,16,24

    // ---- prologue: stage tile 0 into buffer 0 ----
    #pragma unroll
    for (int p = 0; p < 4; ++p) {
        const int r = rStage + p * 64;
        copy16_g2l(A + (size_t)(rowBase + r) * K + cStage,
                   &As[0][r][cStage]);
    }
    lds_stage_fence();

    const f32x8 zero = {0.f, 0.f, 0.f, 0.f, 0.f, 0.f, 0.f, 0.f};
    f32x8 acc[2][4] = {{zero, zero, zero, zero}, {zero, zero, zero, zero}};

    for (int kt = 0; kt < KT; ++kt) {
        const int cur = kt & 1;

        // ---- issue async DMA of the NEXT A tile (overlaps WMMA below) ----
        if (kt + 1 < KT) {
            const int kNext = (kt + 1) * TILE_K;
            #pragma unroll
            for (int p = 0; p < 4; ++p) {
                const int r = rStage + p * 64;
                copy16_g2l(A + (size_t)(rowBase + r) * K + kNext + cStage,
                           &As[cur ^ 1][r][cStage]);
            }
        }

        // ---- A fragments: two contiguous 16B LDS reads each ----
        frag_u fa0, fa1;
        {
            const int r0 = wave * 32 + l15;
            fa0.u4[0] = *(const uint4*)&As[cur][r0][kSel];
            fa0.u4[1] = *(const uint4*)&As[cur][r0][kSel + 16];
            fa1.u4[0] = *(const uint4*)&As[cur][r0 + 16][kSel];
            fa1.u4[1] = *(const uint4*)&As[cur][r0 + 16][kSel + 16];
        }

        // ---- B fragments straight from packed weights (L2-resident) ----
        #pragma unroll
        for (int nt = 0; nt < 4; ++nt) {
            const uint4* bp = (const uint4*)(
                Wp + (((size_t)kt * NT + ntBase + nt) * 32 + lane) * 16);
            frag_u fb;
            fb.u4[0] = bp[0];
            fb.u4[1] = bp[1];
            acc[0][nt] = __builtin_amdgcn_wmma_f32_16x16x32_bf16(
                false, fa0.v, false, fb.v, (short)0, acc[0][nt],
                false, false);
            acc[1][nt] = __builtin_amdgcn_wmma_f32_16x16x32_bf16(
                false, fa1.v, false, fb.v, (short)0, acc[1][nt],
                false, false);
        }

        lds_stage_fence();   // next tile resident + all reads of cur done
    }

    // ---- epilogue: C/D layout — VGPR r: lanes0-15 M=r, lanes16-31 M=8+r ----
    const int hsel = lane >> 4;
    #pragma unroll
    for (int m = 0; m < 2; ++m) {
        const int rowS = rowBase + wave * 32 + m * 16 + hsel * 8;
        #pragma unroll
        for (int nt = 0; nt < 4; ++nt) {
            const int col = (ntBase + nt) * 16 + l15;
            const float bv = bias[col];
            #pragma unroll
            for (int r = 0; r < 8; ++r) {
                float v = acc[m][nt][r] + bv;
                if (EPI == 1) v = fmaxf(v, 0.0f);
                if (EPI == 2) v = __sinf(v);
                const size_t idx = (size_t)(rowS + r) * N + col;
                if (OUTBF) ((__bf16*)Cout)[idx] = f2bf(v);
                else       ((float*)Cout)[idx]  = v;
            }
        }
    }
}

// ---------------------------------------------------------------------------
// Pooling statistics. S: (8,128,128,CHP) f32, 240 live channels.
// rm[z,i,c] = mean_b S[z,i,b,c]; cm[z,i,c] = mean_a S[z,a,i,c];
// dg[z,i,c] = S[z,i,i,c].   (rm/cm/dg stride 240)
// ---------------------------------------------------------------------------
__global__ __launch_bounds__(256) void stats_kernel(
    const float* __restrict__ S, float* __restrict__ rm,
    float* __restrict__ cm, float* __restrict__ dg)
{
    const int z = blockIdx.y;   // 0..7
    const int i = blockIdx.x;   // 0..127
    const int c = threadIdx.x;
    if (c >= 240) return;
    const size_t base = (size_t)z * 128 * 128 * CHP;
    float sr = 0.f, sc = 0.f;
    for (int b = 0; b < 128; ++b) {
        sr += S[base + ((size_t)i * 128 + b) * CHP + c];
        sc += S[base + ((size_t)b * 128 + i) * CHP + c];
    }
    const size_t o = ((size_t)z * 128 + i) * 240 + c;
    rm[o] = sr * (1.0f / 128.0f);
    cm[o] = sc * (1.0f / 128.0f);
    dg[o] = S[base + ((size_t)i * 128 + i) * CHP + c];
}

// tm[z,c] = mean_i rm[z,i,c] (== total mean); dm[z,c] = mean_i dg[z,i,c].
__global__ __launch_bounds__(256) void stats2_kernel(
    const float* __restrict__ rm, const float* __restrict__ dg,
    float* __restrict__ tm, float* __restrict__ dm)
{
    const int z = blockIdx.x;
    const int c = threadIdx.x;
    if (c >= 240) return;
    float st = 0.f, sd = 0.f;
    for (int i = 0; i < 128; ++i) {
        st += rm[((size_t)z * 128 + i) * 240 + c];
        sd += dg[((size_t)z * 128 + i) * 240 + c];
    }
    tm[(size_t)z * 240 + c] = st * (1.0f / 128.0f);
    dm[(size_t)z * 240 + c] = sd * (1.0f / 128.0f);
}

// ---------------------------------------------------------------------------
// Equivariant pooling scatter. Channel c -> head t = c % 15 (k*15+t == c, so
// input/output channel index is identity). Optionally adds skip and writes
// both f32 (for later residual) and bf16 (next GEMM input). Threads 240..255
// zero the channel pad so the next GEMM's K=256 reads clean zeros.
// ---------------------------------------------------------------------------
template <bool HAS_SKIP, bool WRITE_F32>
__global__ __launch_bounds__(256) void pool_kernel(
    const float* __restrict__ S, const float* __restrict__ rm,
    const float* __restrict__ cm, const float* __restrict__ dg,
    const float* __restrict__ tm, const float* __restrict__ dm,
    const float* __restrict__ skip, float* __restrict__ poolf,
    __bf16* __restrict__ poolb)
{
    const int z = blockIdx.y;   // 0..7
    const int a = blockIdx.x;   // 0..127
    const int c = threadIdx.x;
    const size_t zi = (size_t)z * 128;
    if (c >= 240) {             // zero the pad channels
        for (int b = 0; b < 128; ++b) {
            const size_t o = ((zi + a) * 128 + b) * CHP + c;
            if (WRITE_F32) poolf[o] = 0.0f;
            poolb[o] = f2bf(0.0f);
        }
        return;
    }
    const int t = c % 15;
    const float dga = dg[(zi + a) * 240 + c];
    const float rma = rm[(zi + a) * 240 + c];
    const float cma = cm[(zi + a) * 240 + c];
    const float tmv = tm[(size_t)z * 240 + c];
    const float dmv = dm[(size_t)z * 240 + c];
    for (int b = 0; b < 128; ++b) {
        float v;
        switch (t) {
            case 0:  v = S[((zi + a) * 128 + b) * CHP + c]; break; // identity
            case 1:  v = S[((zi + b) * 128 + a) * CHP + c]; break; // transpose
            case 2:  v = dga;                      break; // diag -> rows
            case 3:  v = dg[(zi + b) * 240 + c];   break; // diag -> cols
            case 4:  v = rma;                      break; // row-mean -> rows
            case 5:  v = cma;                      break; // col-mean -> rows
            case 6:  v = rm[(zi + b) * 240 + c];   break; // row-mean -> cols
            case 7:  v = cm[(zi + b) * 240 + c];   break; // col-mean -> cols
            case 8:  v = tmv;                      break; // total mean
            case 9:  v = dmv;                      break; // diag mean
            case 10: v = (a == b) ? dga : 0.f;     break;
            case 11: v = (a == b) ? rma : 0.f;     break;
            case 12: v = (a == b) ? cma : 0.f;     break;
            case 13: v = (a == b) ? tmv : 0.f;     break;
            default: v = (a == b) ? dmv : 0.f;     break;
        }
        const size_t o = ((zi + a) * 128 + b) * CHP + c;
        if (HAS_SKIP)  v += skip[o];
        if (WRITE_F32) poolf[o] = v;
        poolb[o] = f2bf(v);
    }
}

// ---------------------------------------------------------------------------
// Host-side launch
// ---------------------------------------------------------------------------
extern "C" void kernel_launch(void* const* d_in, const int* in_sizes, int n_in,
                              void* d_out, int out_size, void* d_ws,
                              size_t ws_size, hipStream_t stream)
{
    (void)in_sizes; (void)n_in; (void)out_size; (void)ws_size;

    // True dims (Ktrue, Ntrue) and padded dims (Kp, Np) per layer.
    static const int wt[9][2] = {
        {128, 512}, {512, 512}, {512, 240},   // stack 0
        {240, 512}, {512, 512}, {512, 240},   // stack 1
        {240, 512}, {512, 512}, {512, 128}};  // final
    static const int wp[9][2] = {
        {128, 512}, {512, 512}, {512, 256},
        {256, 512}, {512, 512}, {512, 256},
        {256, 512}, {512, 512}, {512, 128}};

    const float* X = (const float*)d_in[0];
    const size_t M = (size_t)8 * 128 * 128;   // 131072 rows
    auto al = [](size_t x) { return (x + 255) & ~(size_t)255; };

    // Packed weight slots (Kp * Np bf16 each) and padded-bias slots.
    size_t woff[9], wtot = 0, boff[9], btot = 0;
    for (int j = 0; j < 9; ++j) {
        woff[j] = wtot; wtot += (size_t)wp[j][0] * wp[j][1];
        boff[j] = btot; btot += (size_t)wp[j][1];
    }

    char* ws = (char*)d_ws;
    size_t off = 0;
    __bf16* Wp  = (__bf16*)(ws + off); off += al(wtot * 2);
    float*  Bp  = (float*)(ws + off);  off += al(btot * 4);
    __bf16* INb = (__bf16*)(ws + off); off += al(M * CHP * 2);
    __bf16* H1  = (__bf16*)(ws + off); off += al(M * 512 * 2);
    __bf16* H2  = (__bf16*)(ws + off); off += al(M * 512 * 2);
    float*  S   = (float*)(ws + off);  off += al(M * CHP * 4);
    float*  P0  = (float*)(ws + off);  off += al(M * CHP * 4);
    float*  RM  = (float*)(ws + off);  off += al((size_t)8 * 128 * 240 * 4);
    float*  CM  = (float*)(ws + off);  off += al((size_t)8 * 128 * 240 * 4);
    float*  DG  = (float*)(ws + off);  off += al((size_t)8 * 128 * 240 * 4);
    float*  TM  = (float*)(ws + off);  off += al((size_t)8 * 240 * 4);
    float*  DM  = (float*)(ws + off);  off += al((size_t)8 * 240 * 4);

    // --- repack weights (f32 -> fragment-major bf16, zero padded), biases ---
    for (int j = 0; j < 9; ++j) {
        const size_t n = (size_t)wp[j][0] * wp[j][1];
        repack_w_kernel<<<(unsigned)((n + 255) / 256), 256, 0, stream>>>(
            (const float*)d_in[1 + 2 * j], Wp + woff[j],
            wt[j][0], wt[j][1], wp[j][1], wp[j][0] / TILE_K);
        pad_bias_kernel<<<(unsigned)((wp[j][1] + 255) / 256), 256, 0, stream>>>(
            (const float*)d_in[2 + 2 * j], Bp + boff[j], wt[j][1], wp[j][1]);
    }
    f32_to_bf16_kernel<<<(unsigned)((M * 128 + 255) / 256), 256, 0, stream>>>(
        X, INb, M * 128);

    const dim3 blk(256);
    const unsigned gm = (unsigned)(M / TILE_M);   // 512

    // ---------------- Stack 0 ----------------
    gemm_wmma_kernel<128, 512, 1, true><<<dim3(gm, 8), blk, 0, stream>>>(
        INb, Wp + woff[0], Bp + boff[0], H1);
    gemm_wmma_kernel<512, 512, 1, true><<<dim3(gm, 8), blk, 0, stream>>>(
        H1, Wp + woff[1], Bp + boff[1], H2);
    gemm_wmma_kernel<512, 256, 2, false><<<dim3(gm, 4), blk, 0, stream>>>(
        H2, Wp + woff[2], Bp + boff[2], S);
    stats_kernel<<<dim3(128, 8), 256, 0, stream>>>(S, RM, CM, DG);
    stats2_kernel<<<8, 256, 0, stream>>>(RM, DG, TM, DM);
    pool_kernel<false, true><<<dim3(128, 8), 256, 0, stream>>>(
        S, RM, CM, DG, TM, DM, nullptr, P0, INb);

    // ---------------- Stack 1 ----------------
    gemm_wmma_kernel<256, 512, 1, true><<<dim3(gm, 8), blk, 0, stream>>>(
        INb, Wp + woff[3], Bp + boff[3], H1);
    gemm_wmma_kernel<512, 512, 1, true><<<dim3(gm, 8), blk, 0, stream>>>(
        H1, Wp + woff[4], Bp + boff[4], H2);
    gemm_wmma_kernel<512, 256, 2, false><<<dim3(gm, 4), blk, 0, stream>>>(
        H2, Wp + woff[5], Bp + boff[5], S);
    stats_kernel<<<dim3(128, 8), 256, 0, stream>>>(S, RM, CM, DG);
    stats2_kernel<<<8, 256, 0, stream>>>(RM, DG, TM, DM);
    pool_kernel<true, false><<<dim3(128, 8), 256, 0, stream>>>(
        S, RM, CM, DG, TM, DM, P0, P0, INb);   // adds skip, bf16 out

    // ---------------- Final MLP ----------------
    gemm_wmma_kernel<256, 512, 1, true><<<dim3(gm, 8), blk, 0, stream>>>(
        INb, Wp + woff[6], Bp + boff[6], H1);
    gemm_wmma_kernel<512, 512, 1, true><<<dim3(gm, 8), blk, 0, stream>>>(
        H1, Wp + woff[7], Bp + boff[7], H2);
    gemm_wmma_kernel<512, 128, 0, false><<<dim3(gm, 2), blk, 0, stream>>>(
        H2, Wp + woff[8], Bp + boff[8], d_out);
}
